// BiSSMBlock_87986700026014
// MI455X (gfx1250) — matrix operand order repaired and checked by reading, not compiled, not run on hardware
//
#include <hip/hip_runtime.h>
#include <cstdint>

// ---------------------------------------------------------------------------
// BiSSM (VMamba cross-scan Mamba) block for gfx1250.
//   B=8, L=1024 (32x32), Dm=384, Di=768, N=16, R=24, K_conv=4.
// GEMMs use v_wmma_f32_16x16x32_bf16 (bf16 A/B, f32 acc); scan is a
// register-resident recurrence with LDS-staged per-timestep rows.
// ---------------------------------------------------------------------------

typedef __attribute__((ext_vector_type(16))) __bf16 v16bf;
typedef __attribute__((ext_vector_type(8)))  float  v8f;

#define BATCH   8
#define LSEQ    1024
#define ROWS    (BATCH * LSEQ)     // 8192
#define DM      384
#define DI      768
#define DSTATE  16
#define DTRANK  24
#define XDBLW   (DTRANK + 2 * DSTATE)   // 56

__device__ __forceinline__ unsigned short f2bf(float f) {
    unsigned u = __float_as_uint(f);
    u += 0x7FFFu + ((u >> 16) & 1u);     // round-to-nearest-even
    return (unsigned short)(u >> 16);
}

// Cross-scan permutation. For this set of orderings the gather (build x_d)
// and scatter (write y_d back) index maps are the same function.
__device__ __forceinline__ int perm_l(int dir, int p) {
    switch (dir & 3) {
        case 0:  return p;                                   // H-forward
        case 1:  return (p & ~31) | (31 - (p & 31));         // H-backward
        case 2:  return ((p & 31) << 5) | (p >> 5);          // V-forward
        default: return ((31 - (p & 31)) << 5) | (p >> 5);   // V-backward
    }
}

// ------------------------------ elementwise --------------------------------

__global__ void cast_bf16_k(const float* __restrict__ src,
                            unsigned short* __restrict__ dst, int n) {
    int i = blockIdx.x * blockDim.x + threadIdx.x;
    if (i < n) dst[i] = f2bf(src[i]);
}

__global__ void permute_cast_x_k(const float* __restrict__ x,
                                 unsigned short* __restrict__ xs, int dir) {
    int idx = blockIdx.x * blockDim.x + threadIdx.x;
    if (idx >= ROWS * DM) return;
    int c = idx % DM, m = idx / DM;
    int b = m >> 10, lp = m & 1023;
    int l = perm_l(dir, lp);
    xs[idx] = f2bf(x[((size_t)(b * LSEQ + l)) * DM + c]);
}

// ------------------------- WMMA bf16 GEMM (C = A * W^T) --------------------
// A: (M,K) bf16 row-major.  W: (N,K) bf16 row-major.  Per-wave 32x32 tile.
// mode 0: C fp32 (ldc)                     [in_proj -> uz]
// mode 1: bf16 store into cat with row permutation + column offset
// mode 2: C fp32 = acc + bias[n] + resid   [final projection -> rbuf]

union AB32 { uint4 q[2]; v16bf v; };

__global__ void gemm_wmma_bf16_k(const unsigned short* __restrict__ A,
                                 const unsigned short* __restrict__ W,
                                 float* __restrict__ C,
                                 unsigned short* __restrict__ Cbf,
                                 const float* __restrict__ bias,
                                 const float* __restrict__ resid,
                                 int M, int N, int K, int ldc,
                                 int mode, int dir, int dcol) {
    const int lane   = threadIdx.x & 31;
    const int wid    = blockIdx.x * (blockDim.x >> 5) + (threadIdx.x >> 5);
    const int tilesM = M >> 5;
    if (wid >= tilesM * (N >> 5)) return;        // uniform per wave
    const int tm = wid % tilesM;
    const int tn = wid / tilesM;

    const int mrow = lane & 15;
    const int half = lane >> 4;

    const size_t rowA0 = (size_t)(tm * 32 + mrow);
    const size_t rowA1 = rowA0 + 16;
    const int    kbA   = half * 8;               // A: K = kb..kb+7, 16+kb..16+kb+7
    const size_t colB0 = (size_t)(tn * 32 + mrow);
    const size_t colB1 = colB0 + 16;
    const int    krB   = half * 8;               // B: K = krB..krB+7, 16+krB..16+krB+7

    v8f acc00 = {0,0,0,0,0,0,0,0};
    v8f acc01 = acc00, acc10 = acc00, acc11 = acc00;

    for (int k0 = 0; k0 < K; k0 += 32) {
        AB32 a0, a1, b0, b1;
        const unsigned short* pa0 = A + rowA0 * K + k0 + kbA;
        const unsigned short* pa1 = A + rowA1 * K + k0 + kbA;
        a0.q[0] = *(const uint4*)(pa0);  a0.q[1] = *(const uint4*)(pa0 + 16);
        a1.q[0] = *(const uint4*)(pa1);  a1.q[1] = *(const uint4*)(pa1 + 16);
        __builtin_prefetch(pa0 + 32, 0, 0);      // next K-chunk of streamed A
        const unsigned short* pb0 = W + colB0 * K + k0 + krB;
        const unsigned short* pb1 = W + colB1 * K + k0 + krB;
        b0.q[0] = *(const uint4*)(pb0);  b0.q[1] = *(const uint4*)(pb0 + 16);
        b1.q[0] = *(const uint4*)(pb1);  b1.q[1] = *(const uint4*)(pb1 + 16);

        acc00 = __builtin_amdgcn_wmma_f32_16x16x32_bf16(false, a0.v, false, b0.v,
                                                        (short)0, acc00, false, false);
        acc01 = __builtin_amdgcn_wmma_f32_16x16x32_bf16(false, a0.v, false, b1.v,
                                                        (short)0, acc01, false, false);
        acc10 = __builtin_amdgcn_wmma_f32_16x16x32_bf16(false, a1.v, false, b0.v,
                                                        (short)0, acc10, false, false);
        acc11 = __builtin_amdgcn_wmma_f32_16x16x32_bf16(false, a1.v, false, b1.v,
                                                        (short)0, acc11, false, false);
    }

    // D layout: VGPR r -> row M = 8*half + r, col N = lane%16.
    #pragma unroll
    for (int t = 0; t < 4; t++) {
        const int ti = t >> 1, tj = t & 1;
        v8f acc = (t == 0) ? acc00 : (t == 1) ? acc01 : (t == 2) ? acc10 : acc11;
        const int n     = tn * 32 + tj * 16 + mrow;
        const int rbase = tm * 32 + ti * 16 + half * 8;
        #pragma unroll
        for (int r = 0; r < 8; r++) {
            const float v  = acc[r];
            const int  row = rbase + r;
            if (mode == 0) {
                C[(size_t)row * ldc + n] = v;
            } else if (mode == 1) {
                int b = row >> 10, lp = row & 1023;
                int m = perm_l(dir, lp);
                Cbf[((size_t)(b * LSEQ + m)) * (4 * DM) + dcol + n] = f2bf(v);
            } else {
                C[(size_t)row * ldc + n] =
                    v + bias[n] + resid[(size_t)row * ldc + n];
            }
        }
    }
}

// ------------------------- conv (depthwise, causal, K=4) + SiLU ------------

__global__ void conv_silu_k(const float* __restrict__ uz,
                            const float* __restrict__ conv_w,
                            const float* __restrict__ conv_b,
                            float* __restrict__ uc, int dir) {
    int idx = blockIdx.x * blockDim.x + threadIdx.x;
    if (idx >= ROWS * DI) return;
    int ch = idx % DI, m = idx / DI;
    int b = m >> 10, l = m & 1023;
    const float* w = conv_w + ((size_t)dir * DI + ch) * 4;
    float acc = conv_b[dir * DI + ch];
    #pragma unroll
    for (int k = 0; k < 4; k++) {
        int ll = l - 3 + k;
        if (ll >= 0)
            acc = fmaf(uz[((size_t)(b * LSEQ + ll)) * (2 * DI) + ch], w[k], acc);
    }
    uc[idx] = acc / (1.0f + __expf(-acc));       // SiLU
}

// ------------------------- x_dbl = uc @ W_x^T (N=56, small) ----------------

__global__ void xdbl_k(const float* __restrict__ uc,
                       const float* __restrict__ W_x,
                       float* __restrict__ xdbl, int dir) {
    int idx = blockIdx.x * blockDim.x + threadIdx.x;
    if (idx >= ROWS * XDBLW) return;
    int n = idx % XDBLW, m = idx / XDBLW;
    const float* wr = W_x + ((size_t)dir * XDBLW + n) * DI;
    const float* ur = uc + (size_t)m * DI;
    float acc = 0.f;
    for (int k = 0; k < DI; k++) acc = fmaf(ur[k], wr[k], acc);
    xdbl[idx] = acc;
}

// ------------------- selective scan + gate (fused delta & SiLU) ------------
// One lane owns one channel: h[16] lives in VGPRs. dt/B/C rows staged in LDS
// in 16-step chunks. grid = (BATCH * DI/256), block = 256.

__global__ void scan_gate_k(const float* __restrict__ xdbl,
                            const float* __restrict__ uc,
                            const float* __restrict__ uz,
                            const float* __restrict__ A_log,
                            const float* __restrict__ W_dt,
                            const float* __restrict__ b_dt,
                            const float* __restrict__ Dpar,
                            unsigned short* __restrict__ ygate, int dir) {
    __shared__ float lds[16 * XDBLW];
    const int tid = threadIdx.x;                 // 256
    const int b   = blockIdx.x / 3;
    const int ch  = (blockIdx.x % 3) * 256 + tid;
    const int gch = dir * DI + ch;

    float Arow[DSTATE], h[DSTATE], Wdt[DTRANK];
    #pragma unroll
    for (int n = 0; n < DSTATE; n++) {
        Arow[n] = -__expf(A_log[(size_t)gch * DSTATE + n]);   // A = -exp(A_log)
        h[n] = 0.f;
    }
    #pragma unroll
    for (int r = 0; r < DTRANK; r++) Wdt[r] = W_dt[(size_t)gch * DTRANK + r];
    const float bdt = b_dt[gch], Dp = Dpar[gch];
    const size_t rowb = (size_t)b * LSEQ;

    for (int c0 = 0; c0 < LSEQ; c0 += 16) {
        __syncthreads();
        for (int t = tid; t < 16 * XDBLW; t += 256)
            lds[t] = xdbl[(rowb + c0) * XDBLW + t];
        __syncthreads();
        for (int s = 0; s < 16; s++) {
            const float* rw = &lds[s * XDBLW];
            float dt = bdt;
            #pragma unroll
            for (int r = 0; r < DTRANK; r++) dt = fmaf(rw[r], Wdt[r], dt);
            // softplus
            const float delta = fmaxf(dt, 0.f) + log1pf(__expf(-fabsf(dt)));
            const size_t off = rowb + c0 + s;
            const float u  = uc[off * DI + ch];
            const float du = delta * u;
            float y = 0.f;
            #pragma unroll
            for (int n = 0; n < DSTATE; n++) {
                h[n] = __expf(delta * Arow[n]) * h[n] + du * rw[DTRANK + n];
                y = fmaf(h[n], rw[DTRANK + DSTATE + n], y);
            }
            const float z  = uz[off * (2 * DI) + DI + ch];
            const float yg = (y + u * Dp) * (z / (1.0f + __expf(-z)));
            ygate[off * DI + ch] = f2bf(yg);
        }
    }
}

// ------------------------------- LayerNorm ---------------------------------

__global__ void layernorm_k(const float* __restrict__ r,
                            const float* __restrict__ g,
                            const float* __restrict__ be,
                            float* __restrict__ out) {
    __shared__ float sm[128];
    const int row = blockIdx.x, tid = threadIdx.x;
    const float* rp = r + (size_t)row * DM;
    float s = 0.f;
    for (int c = tid; c < DM; c += 128) s += rp[c];
    sm[tid] = s; __syncthreads();
    for (int o = 64; o > 0; o >>= 1) { if (tid < o) sm[tid] += sm[tid + o]; __syncthreads(); }
    const float mu = sm[0] / (float)DM; __syncthreads();
    float q = 0.f;
    for (int c = tid; c < DM; c += 128) { float d = rp[c] - mu; q += d * d; }
    sm[tid] = q; __syncthreads();
    for (int o = 64; o > 0; o >>= 1) { if (tid < o) sm[tid] += sm[tid + o]; __syncthreads(); }
    const float inv = rsqrtf(sm[0] / (float)DM + 1e-5f);
    for (int c = tid; c < DM; c += 128)
        out[(size_t)row * DM + c] = g[c] * (rp[c] - mu) * inv + be[c];
}

// ------------------------------- launcher ----------------------------------

extern "C" void kernel_launch(void* const* d_in, const int* in_sizes, int n_in,
                              void* d_out, int out_size, void* d_ws, size_t ws_size,
                              hipStream_t stream) {
    (void)in_sizes; (void)n_in; (void)out_size; (void)ws_size;
    const float* x       = (const float*)d_in[0];
    const float* W_in    = (const float*)d_in[1];
    const float* conv_w  = (const float*)d_in[2];
    const float* conv_b  = (const float*)d_in[3];
    const float* W_x     = (const float*)d_in[4];
    const float* W_dt    = (const float*)d_in[5];
    const float* b_dt    = (const float*)d_in[6];
    const float* A_log   = (const float*)d_in[7];
    const float* D_param = (const float*)d_in[8];
    const float* W_out   = (const float*)d_in[9];
    const float* W_proj  = (const float*)d_in[10];
    const float* b_proj  = (const float*)d_in[11];
    const float* ln_g    = (const float*)d_in[12];
    const float* ln_b    = (const float*)d_in[13];

    char* ws = (char*)d_ws;
    size_t off = 0;
    auto alloc = [&](size_t bytes) -> void* {
        void* p = ws + off; off += (bytes + 255) & ~(size_t)255; return p;
    };
    unsigned short* Win_bf   = (unsigned short*)alloc((size_t)4 * 2 * DI * DM * 2);
    unsigned short* Wout_bf  = (unsigned short*)alloc((size_t)4 * DM * DI * 2);
    unsigned short* Wproj_bf = (unsigned short*)alloc((size_t)DM * 4 * DM * 2);
    unsigned short* xs_bf    = (unsigned short*)alloc((size_t)ROWS * DM * 2);
    float*          uz       = (float*)alloc((size_t)ROWS * 2 * DI * 4);
    float*          uc       = (float*)alloc((size_t)ROWS * DI * 4);
    float*          xdbl     = (float*)alloc((size_t)ROWS * XDBLW * 4);
    unsigned short* ygate    = (unsigned short*)alloc((size_t)ROWS * DI * 2);
    unsigned short* cat_bf   = (unsigned short*)alloc((size_t)ROWS * 4 * DM * 2);
    float*          rbuf     = (float*)alloc((size_t)ROWS * DM * 4);

    // Weight casts (once per call).
    {
        int n1 = 4 * 2 * DI * DM, n2 = 4 * DM * DI, n3 = DM * 4 * DM;
        cast_bf16_k<<<(n1 + 255) / 256, 256, 0, stream>>>(W_in,   Win_bf,   n1);
        cast_bf16_k<<<(n2 + 255) / 256, 256, 0, stream>>>(W_out,  Wout_bf,  n2);
        cast_bf16_k<<<(n3 + 255) / 256, 256, 0, stream>>>(W_proj, Wproj_bf, n3);
    }

    for (int d = 0; d < 4; d++) {
        permute_cast_x_k<<<(ROWS * DM) / 256, 256, 0, stream>>>(x, xs_bf, d);

        // in_proj: (8192 x 1536) = xs (8192x384) @ W_in[d]^T ; tiles 256x48
        gemm_wmma_bf16_k<<<(256 * 48) / 8, 256, 0, stream>>>(
            xs_bf, Win_bf + (size_t)d * 2 * DI * DM,
            uz, nullptr, nullptr, nullptr,
            ROWS, 2 * DI, DM, 2 * DI, /*mode*/0, d, 0);

        conv_silu_k<<<(ROWS * DI) / 256, 256, 0, stream>>>(uz, conv_w, conv_b, uc, d);
        xdbl_k<<<(ROWS * XDBLW) / 256, 256, 0, stream>>>(uc, W_x, xdbl, d);
        scan_gate_k<<<BATCH * 3, 256, 0, stream>>>(
            xdbl, uc, uz, A_log, W_dt, b_dt, D_param, ygate, d);

        // out_proj: (8192 x 384) = ygate (8192x768) @ W_out[d]^T, scattered
        // (row-permuted, bf16) into cat[:, d*384 : (d+1)*384]; tiles 256x12
        gemm_wmma_bf16_k<<<(256 * 12) / 8, 256, 0, stream>>>(
            ygate, Wout_bf + (size_t)d * DM * DI,
            nullptr, cat_bf, nullptr, nullptr,
            ROWS, DM, DI, 0, /*mode*/1, d, d * DM);
    }

    // Final projection: r = cat @ W_proj^T + b_proj + x ; tiles 256x12
    gemm_wmma_bf16_k<<<(256 * 12) / 8, 256, 0, stream>>>(
        cat_bf, Wproj_bf, rbuf, nullptr, b_proj, x,
        ROWS, DM, 4 * DM, DM, /*mode*/2, 0, 0);

    layernorm_k<<<ROWS, 128, 0, stream>>>(rbuf, ln_g, ln_b, (float*)d_out);
}
